// TransformerXLBlock_10746008175398
// MI455X (gfx1250) — compile-verified, matrix operand order checked by reading, not run on hardware
//
#include <hip/hip_runtime.h>
#include <hip/hip_fp16.h>

// ---------------------------------------------------------------------------
// Transformer-XL block for MI455X (gfx1250, wave32, WMMA).
// All GEMMs run through v_wmma_f32_16x16x32_f16 (f32 accumulate).
// ~60 GFLOP of dense math vs 268MB prob output -> HBM-bound; f16 matrix math
// is free accuracy-wise and keeps the matrix pipes saturated.
// ---------------------------------------------------------------------------

#define QLEN   1024
#define BATCH  4
#define DMODEL 512
#define NH     8
#define DH     64
#define MEMLEN 1024
#define KLENX  2048          // MEM + QLEN
#define DMLP   2048
#define QKVD   (3*DMODEL)    // 1536

typedef __attribute__((ext_vector_type(16))) _Float16 v16h;
typedef __attribute__((ext_vector_type(8)))  float    v8f;

__device__ __forceinline__ v8f wmma16(v16h a, v16h b, v8f c) {
  // D = A(16x32 f16) * B(32x16 f16) + C(16x16 f32)
  return __builtin_amdgcn_wmma_f32_16x16x32_f16(false, a, false, b,
                                                (short)0, c, false, false);
}

// Load 16 consecutive f32 (64B-aligned) -> f16 fragment half-row.
// Works for both global and LDS pointers (generic addrspace).
__device__ __forceinline__ v16h load_frag16(const float* p) {
  const float4* p4 = (const float4*)p;
  float4 x0 = p4[0], x1 = p4[1], x2 = p4[2], x3 = p4[3];
  v16h r;
  r[0]=(_Float16)x0.x;  r[1]=(_Float16)x0.y;  r[2]=(_Float16)x0.z;  r[3]=(_Float16)x0.w;
  r[4]=(_Float16)x1.x;  r[5]=(_Float16)x1.y;  r[6]=(_Float16)x1.z;  r[7]=(_Float16)x1.w;
  r[8]=(_Float16)x2.x;  r[9]=(_Float16)x2.y;  r[10]=(_Float16)x2.z; r[11]=(_Float16)x2.w;
  r[12]=(_Float16)x3.x; r[13]=(_Float16)x3.y; r[14]=(_Float16)x3.z; r[15]=(_Float16)x3.w;
  return r;
}

// ---------------------------------------------------------------------------
// C[M,N] = A[M,K] @ B[N,K]^T (+bias, +relu). One wave per 16x64 strip
// (A fragment reused across 4 N-tiles, 4 back-to-back WMMAs per k-step).
// 4 waves per block -> block covers 16x256 of C. Grid: (N/256, M/16).
// ---------------------------------------------------------------------------
template <int FUSE>   // 0 none, 1 +bias, 2 +bias+relu
__global__ void gemm16_f16(const float* __restrict__ A, const float* __restrict__ Bm,
                           const float* __restrict__ bias, float* __restrict__ C,
                           int N, int K) {
  const int lane = threadIdx.x & 31;
  const int wv   = threadIdx.x >> 5;
  const int r16  = lane & 15;
  const int hi   = lane >> 4;
  const int m0   = blockIdx.y * 16;
  const int nb   = (blockIdx.x * 4 + wv) * 64;   // wave's 64-column strip

  const float* ap = A  + (size_t)(m0 + r16) * K + hi * 16;
  const float* bp = Bm + (size_t)(nb + r16) * K + hi * 16;
  const size_t bstep = (size_t)16 * K;           // next N-tile of B rows

  v8f acc0 = {}, acc1 = {}, acc2 = {}, acc3 = {};
  for (int k0 = 0; k0 < K; k0 += 32) {
    __builtin_prefetch(ap + k0 + 64, 0, 0);      // global_prefetch_b8
    __builtin_prefetch(bp + k0 + 64, 0, 0);
    v16h a  = load_frag16(ap + k0);
    v16h b0 = load_frag16(bp + k0);
    v16h b1 = load_frag16(bp + bstep     + k0);
    v16h b2 = load_frag16(bp + bstep * 2 + k0);
    v16h b3 = load_frag16(bp + bstep * 3 + k0);
    acc0 = wmma16(a, b0, acc0);
    acc1 = wmma16(a, b1, acc1);
    acc2 = wmma16(a, b2, acc2);
    acc3 = wmma16(a, b3, acc3);
  }

  const v8f accs[4] = {acc0, acc1, acc2, acc3};
#pragma unroll
  for (int t = 0; t < 4; ++t) {
    const int n = nb + t * 16 + r16;
    float badd = 0.0f;
    if (FUSE >= 1) badd = bias[n];               // hoisted, one load per tile
#pragma unroll
    for (int r = 0; r < 8; ++r) {
      float s = accs[t][r] + badd;
      if (FUSE == 2) s = fmaxf(s, 0.0f);
      C[(size_t)(m0 + r + hi * 8) * N + n] = s;
    }
  }
}

// ---------------------------------------------------------------------------
// V^T staging: Vt[((b*NH+h)*DH+n)*KLENX + j] = V[j,b,h,n]  (f16)
// ---------------------------------------------------------------------------
__global__ void vt_kernel(const float* __restrict__ qkv, _Float16* __restrict__ Vt) {
  size_t idx = (size_t)blockIdx.x * 256 + threadIdx.x;   // 4*8*64*2048 total
  int j    = (int)(idx & (KLENX - 1));
  int rest = (int)(idx >> 11);
  int n    = rest & (DH - 1);
  int bh   = rest >> 6;
  int h    = bh & (NH - 1);
  int b    = bh >> 3;
  Vt[idx] = (_Float16)qkv[((size_t)j * BATCH + b) * QKVD + 2 * DMODEL + h * DH + n];
}

// ---------------------------------------------------------------------------
// Attention: one block per (16 query rows, b, h). 8 waves.
//  score[i,j] = ((q_i+u)·k_j + (q_i+v)·rk[j-i+1023]) * 1/8, causal-masked.
//  Dynamic LDS: sc[16][2048] f32 score strip + per-wave 16x32 BD band + red.
// ---------------------------------------------------------------------------
__global__ void attn_kernel(const float* __restrict__ qkv, const float* __restrict__ rk,
                            const _Float16* __restrict__ Vt,
                            const float* __restrict__ u, const float* __restrict__ v,
                            float* __restrict__ prob_out, float* __restrict__ attn_out) {
  const int i0 = blockIdx.x * 16;
  const int b  = blockIdx.y;
  const int h  = blockIdx.z;

  extern __shared__ float smem[];
  float* sc  = smem;                     // 16 * 2048
  float* bdw = sc + 16 * KLENX;          // 8 waves * 16 * 32
  float* red = bdw + 8 * 16 * 32;        // 16 * 16

  const int lane = threadIdx.x & 31;
  const int wv   = threadIdx.x >> 5;
  const int r16  = lane & 15;
  const int hi   = lane >> 4;

  // ---- A fragments for (q+u) and (q+v), K = DH = 64 -> two 16x32 steps ----
  v16h au0, au1, av0, av1;
  {
    const float* qp = qkv + (((size_t)(MEMLEN + i0 + r16)) * BATCH + b) * QKVD + h * DH + hi * 16;
    const float* up = u + h * DH + hi * 16;
    const float* vp = v + h * DH + hi * 16;
#pragma unroll
    for (int t = 0; t < 16; ++t) {
      float q0 = qp[t], q1 = qp[32 + t];
      au0[t] = (_Float16)(q0 + up[t]);      au1[t] = (_Float16)(q1 + up[32 + t]);
      av0[t] = (_Float16)(q0 + vp[t]);      av1[t] = (_Float16)(q1 + vp[32 + t]);
    }
  }

  // ---- scores: each wave owns 256 key columns (16 tiles of 16) ----
  const float scale = 0.125f;  // 1/sqrt(DH)
  float* bds = bdw + wv * (16 * 32);
  for (int ct = 0; ct < 16; ++ct) {
    const int j0 = wv * 256 + ct * 16;

    // AC = (q+u) @ K^T tile
    const float* krow = qkv + ((size_t)(j0 + r16) * BATCH + b) * QKVD + DMODEL + h * DH + hi * 16;
    v8f ac = {};
    ac = wmma16(au0, load_frag16(krow),      ac);
    ac = wmma16(au1, load_frag16(krow + 32), ac);

    // BD band: l-columns l0..l0+31 (clamped; clamped region is fully masked)
    const int l0 = j0 - i0 + 1008;                 // multiple of 16, >= 0
    const int lA = min(l0 + r16,      KLENX - 1);
    const int lB = min(l0 + 16 + r16, KLENX - 1);
    const float* rA = rk + (size_t)lA * DMODEL + h * DH + hi * 16;
    const float* rB = rk + (size_t)lB * DMODEL + h * DH + hi * 16;
    v8f bd0 = {}, bd1 = {};
    bd0 = wmma16(av0, load_frag16(rA),      bd0);
    bd0 = wmma16(av1, load_frag16(rA + 32), bd0);
    bd1 = wmma16(av0, load_frag16(rB),      bd1);
    bd1 = wmma16(av1, load_frag16(rB + 32), bd1);

#pragma unroll
    for (int r = 0; r < 8; ++r) {
      int m = r + hi * 8;
      bds[m * 32 + r16]      = bd0[r];
      bds[m * 32 + 16 + r16] = bd1[r];
    }
    asm volatile("s_wait_dscnt 0x0" ::: "memory");  // in-wave LDS RAW fence

    // combine AC + shifted BD, mask, stage into score strip
#pragma unroll
    for (int r = 0; r < 8; ++r) {
      int m  = r + hi * 8;
      int i  = i0 + m;
      int j  = j0 + r16;
      int rc = r16 - m + 15;                        // 0..30 band offset
      float s = (ac[r] + bds[m * 32 + rc]) * scale;
      if (j > i + MEMLEN) s = -1e30f;
      sc[m * KLENX + j] = s;
    }
  }
  __syncthreads();

  // ---- softmax over klen; 16 threads per row ----
  {
    const int row = threadIdx.x >> 4;
    const int c16 = threadIdx.x & 15;
    float mx = -1e30f;
    for (int j = c16; j < KLENX; j += 16) mx = fmaxf(mx, sc[row * KLENX + j]);
    red[row * 16 + c16] = mx;
    __syncthreads();
#pragma unroll
    for (int t = 0; t < 16; ++t) mx = fmaxf(mx, red[row * 16 + t]);
    __syncthreads();

    float sum = 0.0f;
    for (int j = c16; j < KLENX; j += 16) {
      float e = __expf(sc[row * KLENX + j] - mx);
      sc[row * KLENX + j] = e;
      sum += e;
    }
    red[row * 16 + c16] = sum;
    __syncthreads();
    sum = 0.0f;
#pragma unroll
    for (int t = 0; t < 16; ++t) sum += red[row * 16 + t];
    float inv = 1.0f / sum;

    const int i = i0 + row;
    for (int j = c16; j < KLENX; j += 16) {
      float p = sc[row * KLENX + j] * inv;
      sc[row * KLENX + j] = p;
      prob_out[(((size_t)i * KLENX + j) * BATCH + b) * NH + h] = p;
    }
  }
  __syncthreads();

  // ---- attn = prob @ V : waves 0..3 produce the 16x64 head output ----
  if (wv < 4) {
    const int n0 = wv * 16;
    const _Float16* vtb = Vt + (((size_t)b * NH + h) * DH + n0 + r16) * KLENX + hi * 16;
    const float* scp = sc + (size_t)r16 * KLENX + hi * 16;
    v8f oacc = {};
    for (int k0 = 0; k0 < KLENX; k0 += 32) {
      v16h a  = load_frag16(scp + k0);             // ds_load_b128 x4 + cvt
      v16h bb = *(const v16h*)(vtb + k0);          // 32B vector load
      oacc = wmma16(a, bb, oacc);
    }
#pragma unroll
    for (int r = 0; r < 8; ++r) {
      int m = r + hi * 8;
      attn_out[((size_t)(i0 + m) * BATCH + b) * DMODEL + h * DH + n0 + r16] = oacc[r];
    }
  }
}

// ---------------------------------------------------------------------------
// out = LayerNorm(x + res) * g + b   (one 256-thread block per 512-col row)
// ---------------------------------------------------------------------------
__global__ void add_ln_kernel(const float* __restrict__ x, const float* __restrict__ res,
                              const float* __restrict__ g, const float* __restrict__ bb,
                              float* __restrict__ out) {
  __shared__ float rbuf[256];
  const int row = blockIdx.x;
  const int t   = threadIdx.x;
  const float* xr = x   + (size_t)row * DMODEL;
  const float* rr = res + (size_t)row * DMODEL;

  float v0 = xr[t] + rr[t];
  float v1 = xr[t + 256] + rr[t + 256];

  rbuf[t] = v0 + v1;
  __syncthreads();
  for (int o = 128; o > 0; o >>= 1) { if (t < o) rbuf[t] += rbuf[t + o]; __syncthreads(); }
  float mu = rbuf[0] * (1.0f / DMODEL);
  __syncthreads();

  float d0 = v0 - mu, d1 = v1 - mu;
  rbuf[t] = d0 * d0 + d1 * d1;
  __syncthreads();
  for (int o = 128; o > 0; o >>= 1) { if (t < o) rbuf[t] += rbuf[t + o]; __syncthreads(); }
  float inv = rsqrtf(rbuf[0] * (1.0f / DMODEL) + 1e-5f);

  out[(size_t)row * DMODEL + t]       = d0 * inv * g[t]       + bb[t];
  out[(size_t)row * DMODEL + t + 256] = d1 * inv * g[t + 256] + bb[t + 256];
}

// ---------------------------------------------------------------------------
extern "C" void kernel_launch(void* const* d_in, const int* in_sizes, int n_in,
                              void* d_out, int out_size, void* d_ws, size_t ws_size,
                              hipStream_t stream) {
  const float* inputs = (const float*)d_in[0];   // [1024,4,512]
  const float* r      = (const float*)d_in[1];   // [2048,512]
  const float* u      = (const float*)d_in[2];   // [8,64]
  const float* v      = (const float*)d_in[3];   // [8,64]
  const float* mem    = (const float*)d_in[4];   // [1024,4,512]
  // d_in[5] attn_mask: recomputed analytically (j > i + MEM)
  const float* Wqkv   = (const float*)d_in[6];   // [1536,512]
  const float* Wr     = (const float*)d_in[7];   // [512,512]
  const float* Wo     = (const float*)d_in[8];   // [512,512]
  const float* ln1g   = (const float*)d_in[9];
  const float* ln1b   = (const float*)d_in[10];
  const float* w1     = (const float*)d_in[11];  // [2048,512]
  const float* b1     = (const float*)d_in[12];
  const float* w2     = (const float*)d_in[13];  // [512,2048]
  const float* b2     = (const float*)d_in[14];
  const float* ln2g   = (const float*)d_in[15];
  const float* ln2b   = (const float*)d_in[16];

  float* out_main = (float*)d_out;                             // [1024,4,512]
  float* prob     = out_main + (size_t)QLEN * BATCH * DMODEL;  // [1024,2048,4,8]

  // workspace layout (~140 MB)
  float* ws    = (float*)d_ws;
  float* cat   = ws;                                   // 8192*512
  float* qkvb  = cat   + (size_t)8192 * 512;           // 8192*1536
  float* rkb   = qkvb  + (size_t)8192 * QKVD;          // 2048*512
  float* attnb = rkb   + (size_t)KLENX * DMODEL;       // 4096*512
  float* aoutb = attnb + (size_t)4096 * DMODEL;        // 4096*512
  float* ybuf  = aoutb + (size_t)4096 * DMODEL;        // 4096*512
  float* hbuf  = ybuf  + (size_t)4096 * DMODEL;        // 4096*2048
  float* zbuf  = hbuf  + (size_t)4096 * DMLP;          // 4096*512
  _Float16* Vt = (_Float16*)(zbuf + (size_t)4096 * DMODEL); // 4*8*64*2048 f16

  const size_t halfBytes = (size_t)MEMLEN * BATCH * DMODEL * sizeof(float); // 8MB

  // cat = [mem; inputs]
  hipMemcpyAsync(cat, mem, halfBytes, hipMemcpyDeviceToDevice, stream);
  hipMemcpyAsync(cat + (size_t)MEMLEN * BATCH * DMODEL, inputs, halfBytes,
                 hipMemcpyDeviceToDevice, stream);

  // qkv = cat @ Wqkv^T   (M=8192, N=1536, K=512)
  gemm16_f16<0><<<dim3(QKVD / 256, 8192 / 16), 128, 0, stream>>>(
      cat, Wqkv, nullptr, qkvb, QKVD, DMODEL);

  // rk = r @ Wr^T        (M=2048, N=512, K=512)
  gemm16_f16<0><<<dim3(DMODEL / 256, KLENX / 16), 128, 0, stream>>>(
      r, Wr, nullptr, rkb, DMODEL, DMODEL);

  // V^T staging
  vt_kernel<<<(BATCH * NH * DH * KLENX) / 256, 256, 0, stream>>>(qkvb, Vt);

  // attention: prob + context
  size_t smemB = (size_t)(16 * KLENX + 8 * 16 * 32 + 16 * 16) * sizeof(float); // ~145KB < 320KB WGP LDS
  attn_kernel<<<dim3(QLEN / 16, BATCH, NH), 256, smemB, stream>>>(
      qkvb, rkb, Vt, u, v, prob, attnb);

  // attn_out = attn @ Wo^T  (M=4096, N=512, K=512)
  gemm16_f16<0><<<dim3(DMODEL / 256, 4096 / 16), 128, 0, stream>>>(
      attnb, Wo, nullptr, aoutb, DMODEL, DMODEL);

  // y = LN(inputs + attn_out)
  add_ln_kernel<<<4096, 256, 0, stream>>>(inputs, aoutb, ln1g, ln1b, ybuf);

  // h = relu(y @ w1^T + b1)  (M=4096, N=2048, K=512)
  gemm16_f16<2><<<dim3(DMLP / 256, 4096 / 16), 128, 0, stream>>>(
      ybuf, w1, b1, hbuf, DMLP, DMODEL);

  // z = h @ w2^T + b2        (M=4096, N=512, K=2048)
  gemm16_f16<1><<<dim3(DMODEL / 256, 4096 / 16), 128, 0, stream>>>(
      hbuf, w2, b2, zbuf, DMODEL, DMLP);

  // out = LN(y + z)
  add_ln_kernel<<<4096, 256, 0, stream>>>(ybuf, zbuf, ln2g, ln2b, out_main);
}
